// DecoderLayer_5935644803101
// MI455X (gfx1250) — compile-verified, matrix operand order checked by reading, not run on hardware
//
#include <hip/hip_runtime.h>
#include <hip/hip_bf16.h>
#include <math.h>

// ---------------- problem constants ----------------
constexpr int BATCH = 2;
constexpr int SEQ   = 1024;
constexpr int DMODEL= 1024;
constexpr int NH    = 8;
constexpr int NKV   = 2;
constexpr int HD    = 128;
constexpr int NEXP  = 8;
constexpr int HID   = 4096;
constexpr int TOK   = BATCH * SEQ;           // 2048 tokens
constexpr float EPS     = 1e-5f;
constexpr float THETA   = 10000.0f;
constexpr float SCALE   = 0.08838834764831845f;
constexpr float SOFTCAP = 30.0f;

// fast hardware tanh on gfx1250 (v_tanh_f32) with safe fallback
#if __has_builtin(__builtin_amdgcn_tanhf)
#define FAST_TANH(x) __builtin_amdgcn_tanhf(x)
#elif __has_builtin(__builtin_amdgcn_tanh_f32)
#define FAST_TANH(x) __builtin_amdgcn_tanh_f32(x)
#else
#define FAST_TANH(x) tanhf(x)
#endif

// ---------------- vector types for WMMA ----------------
typedef __attribute__((ext_vector_type(16))) __bf16 v16bf;
typedef __attribute__((ext_vector_type(8)))  __bf16 v8bf;
typedef __attribute__((ext_vector_type(8)))  float  v8f;

static __device__ inline v16bf make_frag(const __bf16* plo, const __bf16* phi) {
    v8bf lo = *(const v8bf*)plo;
    v8bf hi = *(const v8bf*)phi;
    v16bf r;
#pragma unroll
    for (int i = 0; i < 8; ++i) { r[i] = lo[i]; r[i + 8] = hi[i]; }
    return r;
}

static __device__ inline v8f zero_v8f() {
    v8f z = {0.f, 0.f, 0.f, 0.f, 0.f, 0.f, 0.f, 0.f};
    return z;
}

// =====================================================================
// Generic bf16-WMMA GEMM:  C[M,N](f32) (+)= rowscale(m) * (A[M,K] @ B[K,N])
// A,B are f32 in global; converted to bf16 while staging to LDS.
// LDB is a template constant so B's 16 K-strided staging loads become a
// single clause with immediate offsets off one address.
// Double-buffered LDS; next tile's global loads are issued before this
// tile's WMMAs so memory latency hides under matrix math.
// Block tile 64x128, BK=32, 256 threads = 8 waves (2x4), wave = 32x32.
// =====================================================================
template <int LDB>
__global__ __launch_bounds__(256) void gemm_bf16wmma_kernel(
    const float* __restrict__ A, const float* __restrict__ B,
    float* __restrict__ C,
    int M, int N, int Kd, int lda, int ldc,
    const float* __restrict__ row_scale, int rs_stride, int accumulate)
{
    constexpr int BM = 64, BN = 128, BK = 32, LDT = BK + 8; // 40 bf16 stride (80B)
    __shared__ __align__(16) __bf16 sA[2][BM][LDT];
    __shared__ __align__(16) __bf16 sB[2][BN][LDT];   // N-major: sB[buf][n][k]

    const int tid  = threadIdx.x;
    const int wave = tid >> 5;
    const int lane = tid & 31;
    const int half = lane >> 4;      // 0/1
    const int lr   = lane & 15;
    const int kblo = half * 8;       // lane's low-K base inside a 32-deep fragment

    const int wm = wave >> 2;        // 0..1  (M)
    const int wn = wave & 3;         // 0..3  (N)
    const int m_block = blockIdx.y * BM;
    const int n_block = blockIdx.x * BN;

    // staging geometry
    const int a_row = tid >> 2;            // 0..63
    const int a_kc  = (tid & 3) * 8;       // 0,8,16,24
    const int b_n   = tid & 127;           // 0..127
    const int b_kh  = (tid >> 7) * 16;     // 0 or 16

    float4 ra0, ra1;
    float  rb[16];

    auto load_regs = [&](int k0) {
        const float* ap = &A[(size_t)(m_block + a_row) * lda + k0 + a_kc];
        ra0 = *(const float4*)ap;
        ra1 = *(const float4*)(ap + 4);
        const float* bp = &B[(size_t)(k0 + b_kh) * LDB + n_block + b_n];
#pragma unroll
        for (int i = 0; i < 16; ++i) rb[i] = bp[i * LDB];   // imm-offset clause
    };
    auto store_lds = [&](int buf) {
        v8bf pk;
        pk[0] = (__bf16)ra0.x; pk[1] = (__bf16)ra0.y; pk[2] = (__bf16)ra0.z; pk[3] = (__bf16)ra0.w;
        pk[4] = (__bf16)ra1.x; pk[5] = (__bf16)ra1.y; pk[6] = (__bf16)ra1.z; pk[7] = (__bf16)ra1.w;
        *(v8bf*)&sA[buf][a_row][a_kc] = pk;
        v8bf p0, p1;
#pragma unroll
        for (int i = 0; i < 8; ++i) { p0[i] = (__bf16)rb[i]; p1[i] = (__bf16)rb[i + 8]; }
        *(v8bf*)&sB[buf][b_n][b_kh]     = p0;
        *(v8bf*)&sB[buf][b_n][b_kh + 8] = p1;
    };

    v8f acc[2][2];
#pragma unroll
    for (int i = 0; i < 2; ++i)
#pragma unroll
        for (int j = 0; j < 2; ++j) acc[i][j] = zero_v8f();

    const int ntiles = Kd / BK;
    load_regs(0);
    store_lds(0);
    for (int t = 0; t < ntiles; ++t) {
        __syncthreads();                       // buf[t&1] visible; buf[(t+1)&1] free
        const int buf = t & 1;
        if (t + 1 < ntiles) load_regs((t + 1) * BK);   // overlap with WMMAs below
        if (t + 2 < ntiles) {                  // cache hint two tiles ahead
            __builtin_prefetch(&A[(size_t)(m_block + a_row) * lda + (t + 2) * BK + a_kc], 0, 1);
            __builtin_prefetch(&B[(size_t)((t + 2) * BK + b_kh) * LDB + n_block + b_n], 0, 1);
        }

        v16bf afrag[2], bfrag[2];
#pragma unroll
        for (int tt = 0; tt < 2; ++tt) {
            const __bf16* ap = &sA[buf][wm * 32 + tt * 16 + lr][kblo];
            afrag[tt] = make_frag(ap, ap + 16);
            const __bf16* bp = &sB[buf][wn * 32 + tt * 16 + lr][kblo];
            bfrag[tt] = make_frag(bp, bp + 16);
        }
#pragma unroll
        for (int i = 0; i < 2; ++i)
#pragma unroll
            for (int j = 0; j < 2; ++j)
                acc[i][j] = __builtin_amdgcn_wmma_f32_16x16x32_bf16(
                    false, afrag[i], false, bfrag[j], (short)0, acc[i][j], false, false);

        if (t + 1 < ntiles) store_lds((t + 1) & 1);
    }

    // epilogue: C/D layout -> lane holds col n=lr, rows (half*8 + r)
#pragma unroll
    for (int i = 0; i < 2; ++i)
#pragma unroll
        for (int j = 0; j < 2; ++j) {
            int mbase = m_block + wm * 32 + i * 16 + half * 8;
            int nn    = n_block + wn * 32 + j * 16 + lr;
#pragma unroll
            for (int r = 0; r < 8; ++r) {
                int mg = mbase + r;
                float v = acc[i][j][r];
                if (row_scale) v *= row_scale[(size_t)mg * rs_stride];
                size_t off = (size_t)mg * ldc + nn;
                if (accumulate) C[off] += v; else C[off] = v;
            }
        }
}

// =====================================================================
// RMSNorm: out = (resid?) + rmsnorm(in)*g, one block per token
// =====================================================================
__global__ __launch_bounds__(256) void rmsnorm_kernel(
    const float* __restrict__ in, const float* __restrict__ g,
    const float* __restrict__ resid, float* __restrict__ out, int D)
{
    const int tok = blockIdx.x;
    const float* row = in + (size_t)tok * D;
    float s = 0.f;
    for (int d = threadIdx.x; d < D; d += 256) { float v = row[d]; s += v * v; }
#pragma unroll
    for (int off = 16; off > 0; off >>= 1) s += __shfl_xor(s, off, 32);
    __shared__ float red[8];
    if ((threadIdx.x & 31) == 0) red[threadIdx.x >> 5] = s;
    __syncthreads();
    float tot = 0.f;
#pragma unroll
    for (int i = 0; i < 8; ++i) tot += red[i];
    const float r = rsqrtf(tot / (float)D + EPS);
    float* orow = out + (size_t)tok * D;
    const float* rr = resid ? resid + (size_t)tok * D : nullptr;
    for (int d = threadIdx.x; d < D; d += 256) {
        float v = row[d] * r * g[d];
        orow[d] = rr ? rr[d] + v : v;
    }
}

// =====================================================================
// RoPE + bf16 pack of Q/K, plain bf16 pack of V
// grid: (NH + 2*NKV, SEQ, BATCH), block HD threads
// =====================================================================
__global__ __launch_bounds__(HD) void rope_pack_kernel(
    const float* __restrict__ qkv, __bf16* __restrict__ Qb,
    __bf16* __restrict__ Kb, __bf16* __restrict__ Vb)
{
    const int h = blockIdx.x, s = blockIdx.y, b = blockIdx.z, d = threadIdx.x;
    const float* src = qkv + ((size_t)(b * SEQ + s) * ((NH + 2 * NKV) * HD)) + (size_t)h * HD;
    __bf16* dst;
    bool do_rope = true;
    if (h < NH) {
        dst = Qb + ((size_t)((b * SEQ + s) * NH + h)) * HD;
    } else if (h < NH + NKV) {
        dst = Kb + ((size_t)((b * SEQ + s) * NKV + (h - NH))) * HD;
    } else {
        dst = Vb + ((size_t)((b * SEQ + s) * NKV + (h - NH - NKV))) * HD;
        do_rope = false;
    }
    float v;
    if (do_rope) {
        const int half = HD / 2;
        int i = (d < half) ? d : d - half;
        float freq = __powf(THETA, -(2.0f * (float)i) / (float)HD);
        float ang = (float)s * freq;
        float cs = __cosf(ang), sn = __sinf(ang);
        float x1 = src[i], x2 = src[i + half];
        v = (d < half) ? (x1 * cs - x2 * sn) : (x1 * sn + x2 * cs);
    } else {
        v = src[d];
    }
    dst[d] = (__bf16)v;
}

// =====================================================================
// Flash attention with WMMA (bf16), tanh softcap, causal, GQA.
// grid: (SEQ/64, NH, BATCH), block 128 = 4 waves, wave owns 16 q-rows.
// K tile staged with CDNA5 async global->LDS copies (ASYNCcnt).
// B-fragments preloaded before each WMMA chain so dependent WMMAs
// pipeline back-to-back instead of stalling on ds waits.
// =====================================================================
__global__ __launch_bounds__(128) void attn_kernel(
    const __bf16* __restrict__ Qb, const __bf16* __restrict__ Kb,
    const __bf16* __restrict__ Vb, float* __restrict__ O)
{
    __shared__ __align__(16) __bf16 sK[32][HD];       // [kpos][d]
    __shared__ __align__(16) __bf16 sVt[HD][40];      // [d][kpos] (padded)
    __shared__ __align__(16) __bf16 sP[4][16][40];    // per-wave P tile [q][k]

    const int tid  = threadIdx.x;
    const int wave = tid >> 5;
    const int lane = tid & 31;
    const int half = lane >> 4;
    const int lr   = lane & 15;
    const int kblo = half * 8;

    const int head = blockIdx.y, b = blockIdx.z;
    const int kvh  = head / (NH / NKV);
    const int q0   = blockIdx.x * 64 + wave * 16;

    // Q fragments: 4 chunks of 32 along HD
    v16bf qf[4];
    {
        const __bf16* base = Qb + ((size_t)((b * SEQ + (q0 + lr)) * NH + head)) * HD;
#pragma unroll
        for (int c = 0; c < 4; ++c) {
            const __bf16* p = base + c * 32 + kblo;
            qf[c] = make_frag(p, p + 16);
        }
    }

    v8f acc[8];
#pragma unroll
    for (int i = 0; i < 8; ++i) acc[i] = zero_v8f();
    float mstat[8], lstat[8];
#pragma unroll
    for (int r = 0; r < 8; ++r) { mstat[r] = -3.0e38f; lstat[r] = 0.f; }

    const int kb_end = blockIdx.x * 2 + 2;   // uniform across the block (causal)
    for (int kb = 0; kb < kb_end; ++kb) {
        const int kbase = kb * 32;
        __syncthreads();
        {   // stage K via async global->LDS (bf16, no conversion needed)
            const int kpos = tid >> 2;
            const int dch  = (tid & 3) * 32;
            const __bf16* kg = Kb + ((size_t)((b * SEQ + kbase + kpos) * NKV + kvh)) * HD + dch;
            unsigned lds0 = (unsigned)(size_t)&sK[kpos][dch];
#pragma unroll
            for (int i = 0; i < 4; ++i) {
                const void* src = kg + i * 8;
                unsigned la = lds0 + i * 16;
                asm volatile("global_load_async_to_lds_b128 %0, %1, off"
                             :: "v"(la), "v"(src) : "memory");
            }
            // V staged transposed (scatter) the plain way
            const __bf16* vg = Vb + ((size_t)((b * SEQ + kbase + kpos) * NKV + kvh)) * HD + dch;
#pragma unroll
            for (int i = 0; i < 32; ++i) sVt[dch + i][kpos] = vg[i];
            asm volatile("s_wait_asynccnt 0x0" ::: "memory");
        }
        __syncthreads();

        const bool live = (kbase <= q0 + 15);
        if (live) {
            float st[2][8];
#pragma unroll
            for (int nt = 0; nt < 2; ++nt) {
                // preload all 4 B fragments, then run the dependent WMMA chain
                v16bf bfr[4];
#pragma unroll
                for (int c = 0; c < 4; ++c) {
                    const __bf16* bp = &sK[nt * 16 + lr][c * 32 + kblo];
                    bfr[c] = make_frag(bp, bp + 16);
                }
                v8f s = zero_v8f();
#pragma unroll
                for (int c = 0; c < 4; ++c)
                    s = __builtin_amdgcn_wmma_f32_16x16x32_bf16(
                        false, qf[c], false, bfr[c], (short)0, s, false, false);
#pragma unroll
                for (int r = 0; r < 8; ++r) {
                    float lg = s[r] * SCALE;
                    lg = SOFTCAP * FAST_TANH(lg * (1.0f / SOFTCAP));
                    int qg = q0 + half * 8 + r;
                    int kg = kbase + nt * 16 + lr;
                    st[nt][r] = (kg > qg) ? -1e30f : lg;
                }
            }
            // row max across the 16 lanes of this half + both nt tiles
            float mrow[8];
#pragma unroll
            for (int r = 0; r < 8; ++r) mrow[r] = fmaxf(st[0][r], st[1][r]);
#pragma unroll
            for (int off = 1; off < 16; off <<= 1)
#pragma unroll
                for (int r = 0; r < 8; ++r) mrow[r] = fmaxf(mrow[r], __shfl_xor(mrow[r], off, 32));
            float corr[8], rs[8];
#pragma unroll
            for (int r = 0; r < 8; ++r) {
                float mn = fmaxf(mstat[r], mrow[r]);
                corr[r] = __expf(mstat[r] - mn);
                mstat[r] = mn;
                rs[r] = 0.f;
            }
            // exp, stash P to LDS in [q][k] layout
#pragma unroll
            for (int nt = 0; nt < 2; ++nt)
#pragma unroll
                for (int r = 0; r < 8; ++r) {
                    float p = __expf(st[nt][r] - mstat[r]);
                    rs[r] += p;
                    sP[wave][half * 8 + r][nt * 16 + lr] = (__bf16)p;
                }
#pragma unroll
            for (int off = 1; off < 16; off <<= 1)
#pragma unroll
                for (int r = 0; r < 8; ++r) rs[r] += __shfl_xor(rs[r], off, 32);
#pragma unroll
            for (int r = 0; r < 8; ++r) lstat[r] = lstat[r] * corr[r] + rs[r];
#pragma unroll
            for (int t = 0; t < 8; ++t)
#pragma unroll
                for (int r = 0; r < 8; ++r) acc[t][r] *= corr[r];
        }
        __builtin_amdgcn_wave_barrier();   // keep P stores before P loads (per-wave LDS in order)
        if (live) {
            const __bf16* pp = &sP[wave][lr][kblo];
            v16bf pf = make_frag(pp, pp + 16);
#pragma unroll
            for (int ot = 0; ot < 8; ot += 2) {
                const __bf16* bp0 = &sVt[ot * 16 + lr][kblo];
                const __bf16* bp1 = &sVt[(ot + 1) * 16 + lr][kblo];
                v16bf vf0 = make_frag(bp0, bp0 + 16);
                v16bf vf1 = make_frag(bp1, bp1 + 16);
                acc[ot] = __builtin_amdgcn_wmma_f32_16x16x32_bf16(
                    false, pf, false, vf0, (short)0, acc[ot], false, false);
                acc[ot + 1] = __builtin_amdgcn_wmma_f32_16x16x32_bf16(
                    false, pf, false, vf1, (short)0, acc[ot + 1], false, false);
            }
        }
    }

    // normalize + store O [B,S,NH*HD]
#pragma unroll
    for (int ot = 0; ot < 8; ++ot)
#pragma unroll
        for (int r = 0; r < 8; ++r) {
            int qg = q0 + half * 8 + r;
            int dc = ot * 16 + lr;
            float v = acc[ot][r] / fmaxf(lstat[r], 1e-30f);
            O[((size_t)((b * SEQ + qg) * NH + head)) * HD + dc] = v;
        }
}

// =====================================================================
// Router: logits = h2 @ w_router[D,E]; probs = softmax(softmax(logits));
// weights = (top-2 multi-hot) * probs.  One block (256 thr) per token.
// =====================================================================
__global__ __launch_bounds__(256) void router_kernel(
    const float* __restrict__ h2, const float* __restrict__ w_router,
    float* __restrict__ probs_out, float* __restrict__ weights_out)
{
    const int tok = blockIdx.x;
    const int wave = threadIdx.x >> 5;    // expert index
    const int lane = threadIdx.x & 31;
    float s = 0.f;
    const float* row = h2 + (size_t)tok * DMODEL;
    for (int d = lane; d < DMODEL; d += 32) s += row[d] * w_router[(size_t)d * NEXP + wave];
#pragma unroll
    for (int off = 16; off > 0; off >>= 1) s += __shfl_xor(s, off, 32);
    __shared__ float lg[NEXP];
    if (lane == 0) lg[wave] = s;
    __syncthreads();
    if (threadIdx.x == 0) {
        float t[NEXP], p[NEXP];
        float mx = -3e38f;
#pragma unroll
        for (int e = 0; e < NEXP; ++e) mx = fmaxf(mx, lg[e]);
        float sum = 0.f;
#pragma unroll
        for (int e = 0; e < NEXP; ++e) { t[e] = __expf(lg[e] - mx); sum += t[e]; }
#pragma unroll
        for (int e = 0; e < NEXP; ++e) t[e] /= sum;
        mx = -3e38f;
#pragma unroll
        for (int e = 0; e < NEXP; ++e) mx = fmaxf(mx, t[e]);
        sum = 0.f;
#pragma unroll
        for (int e = 0; e < NEXP; ++e) { p[e] = __expf(t[e] - mx); sum += p[e]; }
#pragma unroll
        for (int e = 0; e < NEXP; ++e) p[e] /= sum;
        // top-2
        int i1 = 0;
#pragma unroll
        for (int e = 1; e < NEXP; ++e) if (p[e] > p[i1]) i1 = e;
        int i2 = (i1 == 0) ? 1 : 0;
#pragma unroll
        for (int e = 0; e < NEXP; ++e) if (e != i1 && p[e] > p[i2]) i2 = e;
#pragma unroll
        for (int e = 0; e < NEXP; ++e) {
            probs_out[(size_t)tok * NEXP + e] = p[e];
            weights_out[(size_t)tok * NEXP + e] = (e == i1 || e == i2) ? p[e] : 0.f;
        }
    }
}

// =====================================================================
// GELU-gate: act[m,i] = gelu(h12[m, HID+i]) * h12[m, i]
// =====================================================================
__global__ __launch_bounds__(256) void glu_kernel(
    const float* __restrict__ h12, float* __restrict__ act)
{
    size_t i = (size_t)blockIdx.x * 256 + threadIdx.x;
    if (i >= (size_t)TOK * HID) return;
    size_t m = i >> 12;            // / HID (4096)
    size_t hd = i & (HID - 1);
    float xe   = h12[m * (2 * HID) + hd];
    float gate = h12[m * (2 * HID) + HID + hd];
    float gl = 0.5f * gate * (1.0f + erff(gate * 0.70710678118654752f));
    act[i] = gl * xe;
}

__global__ __launch_bounds__(256) void zero_kernel(float* __restrict__ p, size_t n) {
    size_t i = (size_t)blockIdx.x * 256 + threadIdx.x;
    if (i < n) p[i] = 0.f;
}

// =====================================================================
// host-side orchestration
// =====================================================================
extern "C" void kernel_launch(void* const* d_in, const int* in_sizes, int n_in,
                              void* d_out, int out_size, void* d_ws, size_t ws_size,
                              hipStream_t stream)
{
    const float* x        = (const float*)d_in[0];
    const float* w_qkv    = (const float*)d_in[1];
    const float* w_o      = (const float*)d_in[2];
    const float* w1       = (const float*)d_in[3];
    const float* w2       = (const float*)d_in[4];
    const float* w_router = (const float*)d_in[5];
    const float* g_pre_mqa  = (const float*)d_in[6];
    const float* g_post_mqa = (const float*)d_in[7];
    const float* g_pre_moe  = (const float*)d_in[8];
    const float* g_post_moe = (const float*)d_in[9];

    float* out_x     = (float*)d_out;
    float* out_probs = out_x + (size_t)TOK * DMODEL;

    // ---- workspace carve-out ----
    char* base = (char*)d_ws;
    size_t off = 0;
    auto carve = [&](size_t bytes) -> char* {
        char* p = base + off;
        off = (off + bytes + 255) & ~(size_t)255;
        return p;
    };
    float*  h    = (float*) carve((size_t)TOK * DMODEL * 4);           // pre-attn norm (reused as h2)
    float*  qkv  = (float*) carve((size_t)TOK * 1536 * 4);             // qkv, reused as attn out
    __bf16* Qb   = (__bf16*)carve((size_t)TOK * NH * HD * 2);
    __bf16* Kb   = (__bf16*)carve((size_t)TOK * NKV * HD * 2);
    __bf16* Vb   = (__bf16*)carve((size_t)TOK * NKV * HD * 2);
    float*  atto = (float*) carve((size_t)TOK * DMODEL * 4);           // o-proj out
    float*  x1   = (float*) carve((size_t)TOK * DMODEL * 4);           // residual after attn
    float*  wts  = (float*) carve((size_t)TOK * NEXP * 4);             // routing weights
    float*  h12  = (float*) carve((size_t)TOK * 2 * HID * 4);          // per-expert w1 out
    float*  act  = (float*) carve((size_t)TOK * HID * 4);              // per-expert activation
    float*  moe  = (float*) carve((size_t)TOK * DMODEL * 4);           // moe accumulator
    if (off > ws_size) return;  // workspace too small; nothing sane to do

    float* attn = qkv;  // attention output reuses qkv buffer (f32 [TOK, NH*HD])
    float* h2   = h;    // pre-MoE norm reuses h

    // 1) h = rmsnorm(x, g_pre_mqa)
    rmsnorm_kernel<<<TOK, 256, 0, stream>>>(x, g_pre_mqa, nullptr, h, DMODEL);
    // 2) qkv = h @ w_qkv   [2048 x 1024 x 1536]
    gemm_bf16wmma_kernel<1536><<<dim3(1536 / 128, TOK / 64), 256, 0, stream>>>(
        h, w_qkv, qkv, TOK, 1536, DMODEL, DMODEL, 1536, nullptr, 0, 0);
    // 3) rope + pack bf16
    rope_pack_kernel<<<dim3(NH + 2 * NKV, SEQ, BATCH), HD, 0, stream>>>(qkv, Qb, Kb, Vb);
    // 4) flash attention -> attn (f32, overwrites qkv buffer)
    attn_kernel<<<dim3(SEQ / 64, NH, BATCH), 128, 0, stream>>>(Qb, Kb, Vb, attn);
    // 5) o-proj
    gemm_bf16wmma_kernel<DMODEL><<<dim3(DMODEL / 128, TOK / 64), 256, 0, stream>>>(
        attn, w_o, atto, TOK, DMODEL, NH * HD, NH * HD, DMODEL, nullptr, 0, 0);
    // 6) x1 = x + rmsnorm(atto, g_post_mqa)
    rmsnorm_kernel<<<TOK, 256, 0, stream>>>(atto, g_post_mqa, x, x1, DMODEL);
    // 7) h2 = rmsnorm(x1, g_pre_moe)
    rmsnorm_kernel<<<TOK, 256, 0, stream>>>(x1, g_pre_moe, nullptr, h2, DMODEL);
    // 8) router -> probs (second output) + routing weights
    router_kernel<<<TOK, 256, 0, stream>>>(h2, w_router, out_probs, wts);
    // 9) zero MoE accumulator
    {
        size_t n = (size_t)TOK * DMODEL;
        zero_kernel<<<(unsigned)((n + 255) / 256), 256, 0, stream>>>(moe, n);
    }
    // 10) dense experts, serialized on the stream so moe += is race-free
    for (int e = 0; e < NEXP; ++e) {
        const float* w1e = w1 + (size_t)e * DMODEL * (2 * HID);
        const float* w2e = w2 + (size_t)e * HID * DMODEL;
        gemm_bf16wmma_kernel<2 * HID><<<dim3((2 * HID) / 128, TOK / 64), 256, 0, stream>>>(
            h2, w1e, h12, TOK, 2 * HID, DMODEL, DMODEL, 2 * HID, nullptr, 0, 0);
        {
            size_t n = (size_t)TOK * HID;
            glu_kernel<<<(unsigned)((n + 255) / 256), 256, 0, stream>>>(h12, act);
        }
        gemm_bf16wmma_kernel<DMODEL><<<dim3(DMODEL / 128, TOK / 64), 256, 0, stream>>>(
            act, w2e, moe, TOK, DMODEL, HID, HID, DMODEL,
            wts + e, NEXP, /*accumulate=*/1);
    }
    // 11) out = x1 + rmsnorm(moe, g_post_moe)
    rmsnorm_kernel<<<TOK, 256, 0, stream>>>(moe, g_post_moe, x1, out_x, DMODEL);
    (void)in_sizes; (void)n_in; (void)out_size;
}
